// ProjEnc_60610578481710
// MI455X (gfx1250) — compile-verified, matrix operand order checked by reading, not run on hardware
//
#include <hip/hip_runtime.h>
#include <hip/hip_bf16.h>

// ProjEnc pipeline for gfx1250 (MI455X), wave32 + WMMA bf16 + b128 loads +
// global_prefetch. Algebraic refactor: g[b,n,k,:] = W1@x[idx] + (W2-W1)@x[n];
// GroupNorm+Leaky are monotonic -> max over k commutes; stats via f32 atomics.

#define BB 4
#define NN 2048
#define KK 32
#define TD 64
#define GD 128
#define OBJ 224
#define NSEG (OBJ*OBJ)

typedef __attribute__((ext_vector_type(16))) __bf16 v16bf;
typedef __attribute__((ext_vector_type(8)))  float  v8f;

__device__ __forceinline__ unsigned short f2bf(float f){
  union { float f; unsigned u; } v; v.f = f;
  unsigned r = v.u + 0x7FFFu + ((v.u >> 16) & 1u);   // round-to-nearest-even
  return (unsigned short)(r >> 16);
}
__device__ __forceinline__ __bf16 us2bf(unsigned short u){
  return __builtin_bit_cast(__bf16, u);
}
__device__ __forceinline__ void atomAddF(float* p, float v){
  __hip_atomic_fetch_add(p, v, __ATOMIC_RELAXED, __HIP_MEMORY_SCOPE_AGENT);
}
// ISA 7.12.2 16-bit A/B fragment K index for element e of lane:
// K = (e/8)*16 + (lane/16)*8 + (e%8)
__device__ __forceinline__ int fragK(int lane, int e){
  return ((e >> 3) << 4) + ((lane >> 4) << 3) + (e & 7);
}

// ---------------------------------------------------------------- fill zero
__global__ void fill_zero_f4(float4* p, long n4){
  long i = (long)blockIdx.x * blockDim.x + threadIdx.x;
  long stride = (long)gridDim.x * blockDim.x;
  float4 z = {0.f, 0.f, 0.f, 0.f};
  for (; i < n4; i += stride) p[i] = z;
}

// -------------------------------------------------- weight fragment packing
// wfU: [ks(2)][ntile(16)][lane(32)][e(16)] bf16, combined U = [W1 | W2-W1]
// wfP: [ks(4)][ntile(8)][lane(32)][e(16)]  bf16, proj weights
__global__ void prep_kernel(const float* __restrict__ wg, const float* __restrict__ wp,
                            unsigned short* wfU, unsigned short* wfP, float* stats){
  int tid = threadIdx.x;                       // 256 threads
  if (tid < BB*8) stats[tid] = 0.f;            // zero group sums/sumsq
  for (int q = tid; q < 2*16*32*16; q += 256){
    int e = q & 15, lane = (q >> 4) & 31, nt = (q >> 9) & 15, ks = q >> 13;
    int n = nt*16 + (lane & 15);
    int k = ks*32 + fragK(lane, e);
    float val;
    if (n < GD) val = wg[n*(2*TD) + k];                         // W1[n][k]
    else { int n2 = n - GD; val = wg[n2*(2*TD) + TD + k] - wg[n2*(2*TD) + k]; }
    wfU[q] = f2bf(val);
  }
  for (int q = tid; q < 4*8*32*16; q += 256){
    int e = q & 15, lane = (q >> 4) & 31, nt = (q >> 9) & 7, ks = q >> 12;
    int n = nt*16 + (lane & 15);
    int k = ks*32 + fragK(lane, e);
    wfP[q] = f2bf(wp[n*GD + k]);
  }
}

// -------------------------------------------------------------- seg indices
__device__ float blockReduceMM(float v, int isMax, float* sm){
  #pragma unroll
  for (int o = 16; o >= 1; o >>= 1){
    float t = __shfl_xor(v, o, 32);
    v = isMax ? fmaxf(v, t) : fminf(v, t);
  }
  if ((threadIdx.x & 31) == 0) sm[threadIdx.x >> 5] = v;
  __syncthreads();
  if (threadIdx.x == 0){
    float r = sm[0];
    for (int i = 1; i < (int)(blockDim.x >> 5); ++i) r = isMax ? fmaxf(r, sm[i]) : fminf(r, sm[i]);
    sm[0] = r;
  }
  __syncthreads();
  float r = sm[0];
  __syncthreads();
  return r;
}

__global__ void seg_kernel(const float* __restrict__ pc, int* __restrict__ seg){
  int b = blockIdx.x, tid = threadIdx.x;       // 256 threads per batch
  __shared__ float sm[8];
  float mnx = 3e38f, mny = 3e38f, mxx = -3e38f, mxy = -3e38f;
  for (int i = tid; i < NN; i += 256){
    float px = pc[(b*NN+i)*3], py = pc[(b*NN+i)*3+1];
    mnx = fminf(mnx, px); mxx = fmaxf(mxx, px);
    mny = fminf(mny, py); mxy = fmaxf(mxy, py);
  }
  mnx = blockReduceMM(mnx, 0, sm);
  mxx = blockReduceMM(mxx, 1, sm);
  mny = blockReduceMM(mny, 0, sm);
  mxy = blockReduceMM(mxy, 1, sm);
  float grid = fmaxf(mxx - mnx, mxy - mny) / (float)(OBJ - 3);
  float imn0 = 3e38f, imn1 = 3e38f, imx0 = -3e38f, imx1 = -3e38f;
  for (int i = tid; i < NN; i += 256){
    float ix = floorf((pc[(b*NN+i)*3]   - mnx) / grid);
    float iy = floorf((pc[(b*NN+i)*3+1] - mny) / grid);
    imn0 = fminf(imn0, ix); imx0 = fmaxf(imx0, ix);
    imn1 = fminf(imn1, iy); imx1 = fmaxf(imx1, iy);
  }
  imn0 = blockReduceMM(imn0, 0, sm);
  imx0 = blockReduceMM(imx0, 1, sm);
  imn1 = blockReduceMM(imn1, 0, sm);
  imx1 = blockReduceMM(imx1, 1, sm);
  // dense = idx_xy + OFFSET + 1; max = imx+2, min = imn; center = floor((max+min)/2)
  float c0 = floorf((imx0 + 2.f + imn0) * 0.5f);
  float c1 = floorf((imx1 + 2.f + imn1) * 0.5f);
  float sh0 = (float)(OBJ/2) - c0;             // (+1 from dense) + (OBJ/2 - c - 1)
  float sh1 = (float)(OBJ/2) - c1;
  for (int i = tid; i < NN; i += 256){
    float ix = floorf((pc[(b*NN+i)*3]   - mnx) / grid);
    float iy = floorf((pc[(b*NN+i)*3+1] - mny) / grid);
    #pragma unroll
    for (int j = 0; j < 9; ++j){
      float d0 = ix + (float)(j/3 - 1) + sh0;
      float d1 = iy + (float)(j%3 - 1) + sh1;
      seg[(b*NN+i)*9 + j] = (int)(d0 * (float)OBJ + d1);
    }
  }
}

// ------------------------------------------------------- input transform x
__global__ void intrans_kernel(const float* __restrict__ opc, const float* __restrict__ w_in,
                               const float* __restrict__ b_in, float* __restrict__ x){
  int bn = blockIdx.x; int o = threadIdx.x;    // 64 threads
  float p0 = opc[bn*3], p1 = opc[bn*3+1], p2 = opc[bn*3+2];
  x[bn*TD + o] = w_in[o*3]*p0 + w_in[o*3+1]*p1 + w_in[o*3+2]*p2 + b_in[o];
}

// ----------------------------------------------------------- kNN (wave/query)
__global__ void knn_kernel(const float* __restrict__ opc, int* __restrict__ idx){
  int blk = blockIdx.x;                        // BB*NN/8 blocks, 8 waves/block
  int b = blk / (NN/8);
  int q0 = (blk % (NN/8)) * 8;
  int tid = threadIdx.x, lane = tid & 31, wave = tid >> 5;
  __shared__ float cp[NN*3];                   // 24 KB candidate stage
  for (int i = tid; i < NN*3; i += 256) cp[i] = opc[(b*NN)*3 + i];
  __syncthreads();
  int q = q0 + wave;
  float qx = cp[q*3], qy = cp[q*3+1], qz = cp[q*3+2];
  float dist[64];                              // 64 candidates per lane, registers
  #pragma unroll
  for (int t = 0; t < 64; ++t){
    int j = lane + 32*t;
    float dx = cp[j*3]-qx, dy = cp[j*3+1]-qy, dz = cp[j*3+2]-qz;
    dist[t] = dx*dx + dy*dy + dz*dz;
  }
  for (int it = 0; it < KK; ++it){
    float m = dist[0]; int tl = 0;
    #pragma unroll
    for (int t = 1; t < 64; ++t) if (dist[t] < m){ m = dist[t]; tl = t; }
    int widx = lane + 32*tl;
    #pragma unroll
    for (int off = 16; off >= 1; off >>= 1){
      float om = __shfl_xor(m, off, 32);
      int   ow = __shfl_xor(widx, off, 32);
      if (om < m || (om == m && ow < widx)){ m = om; widx = ow; }
    }
    if (lane == 0) idx[((b*NN)+q)*KK + it] = widx;
    if ((widx & 31) == lane){                  // owner invalidates (static indices)
      int tt = widx >> 5;
      #pragma unroll
      for (int t = 0; t < 64; ++t) if (t == tt) dist[t] = 3.0e38f;
    }
  }
}

// --------------------------------------- WMMA GEMM: y = x@W1^T, base = x@(W2-W1)^T
__global__ void gemm_yb_kernel(const float* __restrict__ x, const unsigned short* __restrict__ wfU,
                               float* __restrict__ y, float* __restrict__ base){
  int tile = blockIdx.x;                       // BB*NN/16 tiles, 512 thr = 16 waves
  int b = tile / (NN/16);
  int row0 = (tile % (NN/16)) * 16;
  int tid = threadIdx.x, lane = tid & 31, wave = tid >> 5;
  __shared__ unsigned short sA[16*TD];
  // 16x64 f32 A tile = 256 float4; threads 0..255 each load one b128
  if (tid < 256){
    const float4* x4 = (const float4*)(x + ((long)(b*NN)+row0)*TD);
    float4 v = x4[tid];
    int o = tid * 4;
    sA[o+0] = f2bf(v.x); sA[o+1] = f2bf(v.y); sA[o+2] = f2bf(v.z); sA[o+3] = f2bf(v.w);
  }
  __syncthreads();
  v8f acc = {};
  int mrow = lane & 15;
  #pragma unroll
  for (int ks = 0; ks < 2; ++ks){
    v16bf a, w;
    #pragma unroll
    for (int e = 0; e < 16; ++e){
      int k = ks*32 + fragK(lane, e);
      a[e] = us2bf(sA[mrow*TD + k]);
      w[e] = us2bf(wfU[(((ks*16 + wave)*32) + lane)*16 + e]);
    }
    acc = __builtin_amdgcn_wmma_f32_16x16x32_bf16(false, a, false, w, (short)0, acc, false, false);
  }
  int col = wave*16 + mrow;                    // 0..255 combined output
  int rhalf = (lane >> 4) * 8;
  #pragma unroll
  for (int r = 0; r < 8; ++r){
    int n = row0 + r + rhalf;
    float v = acc[r];
    if (col < GD) y[((b*NN)+n)*GD + col] = v;
    else          base[((b*NN)+n)*GD + (col - GD)] = v;
  }
}

// ---------------------------- gather neighbors + running max + group stats
// One wave per point; each lane covers 4 channels (b128 loads); neighbor index
// broadcast via shfl; prefetch 4 neighbors ahead (global_prefetch_b8).
__global__ void gather_max_kernel(const float4* __restrict__ y4, const float4* __restrict__ base4,
                                  const int* __restrict__ idx, float4* __restrict__ m4,
                                  float* __restrict__ stats){
  int lane = threadIdx.x & 31, wave = threadIdx.x >> 5;
  int bn = blockIdx.x * 4 + wave;              // BB*NN/4 blocks of 128 threads
  int b = bn / NN;
  int myidx = idx[bn*KK + lane];               // lane k holds neighbor index k
  float4 bc = base4[(long)bn*32 + lane];
  float4 mv = {-3.0e38f, -3.0e38f, -3.0e38f, -3.0e38f};
  float s = 0.f, s2 = 0.f;
  long rowbase = (long)(b*NN) * 32;
  for (int k = 0; k < KK; ++k){
    int j  = __shfl(myidx, k, 32);
    int jn = __shfl(myidx, (k + 4) & (KK - 1), 32);
    __builtin_prefetch(&y4[rowbase + (long)jn*32 + lane], 0, 1);
    float4 v = y4[rowbase + (long)j*32 + lane];
    v.x += bc.x; v.y += bc.y; v.z += bc.z; v.w += bc.w;
    mv.x = fmaxf(mv.x, v.x); mv.y = fmaxf(mv.y, v.y);
    mv.z = fmaxf(mv.z, v.z); mv.w = fmaxf(mv.w, v.w);
    s  += v.x + v.y + v.z + v.w;
    s2 += v.x*v.x + v.y*v.y + v.z*v.z + v.w*v.w;
  }
  m4[(long)bn*32 + lane] = mv;
  // lanes [g*8, g*8+8) cover channels of group g; reduce within 8-lane groups
  #pragma unroll
  for (int off = 4; off >= 1; off >>= 1){
    s  += __shfl_xor(s,  off, 32);
    s2 += __shfl_xor(s2, off, 32);
  }
  if ((lane & 7) == 0){
    int g = lane >> 3;
    atomAddF(&stats[b*8 + g], s);
    atomAddF(&stats[b*8 + 4 + g], s2);
  }
}

// ------------------------------------------------ per-channel scale/shift
__global__ void finalize_kernel(const float* __restrict__ stats, const float* __restrict__ gamma,
                                const float* __restrict__ beta, float* __restrict__ scale,
                                float* __restrict__ shift){
  int t = threadIdx.x + blockIdx.x * blockDim.x;   // BB*GD = 512
  if (t >= BB*GD) return;
  int b = t / GD, c = t % GD, g = c >> 5;
  float cnt = (float)NN * (float)KK * 32.0f;
  float mean = stats[b*8 + g] / cnt;
  float var  = stats[b*8 + 4 + g] / cnt - mean*mean;
  float rstd = rsqrtf(var + 1e-5f);
  float sc = gamma[c] * rstd;
  scale[t] = sc;
  shift[t] = beta[c] - mean * sc;
}

// --------------------- normalize + LeakyReLU + proj GEMM (WMMA bf16, K=128)
__global__ void norm_proj_kernel(const float4* __restrict__ m4, const float4* __restrict__ scale4,
                                 const float4* __restrict__ shift4, const unsigned short* __restrict__ wfP,
                                 const float* __restrict__ bproj, float* __restrict__ f){
  int tile = blockIdx.x;                       // BB*NN/16 tiles, 256 thr = 8 waves
  int b = tile / (NN/16);
  int row0 = (tile % (NN/16)) * 16;
  int tid = threadIdx.x, lane = tid & 31, wave = tid >> 5;
  __shared__ unsigned short sA[16*GD];
  {                                            // 16x128 tile = 512 float4
    for (int i4 = tid; i4 < 16*GD/4; i4 += 256){
      int c4 = i4 & (GD/4 - 1);                // float4 column within row
      float4 v  = m4[(long)(b*NN + row0)*32 + i4];
      float4 sc = scale4[b*(GD/4) + c4];
      float4 sh = shift4[b*(GD/4) + c4];
      float h0 = v.x*sc.x + sh.x, h1 = v.y*sc.y + sh.y;
      float h2 = v.z*sc.z + sh.z, h3 = v.w*sc.w + sh.w;
      h0 = (h0 >= 0.f) ? h0 : 0.2f*h0;  h1 = (h1 >= 0.f) ? h1 : 0.2f*h1;
      h2 = (h2 >= 0.f) ? h2 : 0.2f*h2;  h3 = (h3 >= 0.f) ? h3 : 0.2f*h3;
      int o = i4 * 4;
      sA[o+0] = f2bf(h0); sA[o+1] = f2bf(h1); sA[o+2] = f2bf(h2); sA[o+3] = f2bf(h3);
    }
  }
  __syncthreads();
  v8f acc = {};
  int mrow = lane & 15;
  #pragma unroll
  for (int ks = 0; ks < 4; ++ks){
    v16bf a, w;
    #pragma unroll
    for (int e = 0; e < 16; ++e){
      int k = ks*32 + fragK(lane, e);
      a[e] = us2bf(sA[mrow*GD + k]);
      w[e] = us2bf(wfP[(((ks*8 + wave)*32) + lane)*16 + e]);
    }
    acc = __builtin_amdgcn_wmma_f32_16x16x32_bf16(false, a, false, w, (short)0, acc, false, false);
  }
  int col = wave*16 + mrow;
  int rhalf = (lane >> 4) * 8;
  float bp = bproj[col];
  #pragma unroll
  for (int r = 0; r < 8; ++r){
    int n = row0 + r + rhalf;
    f[((b*NN)+n)*GD + col] = acc[r] + bp;
  }
}

// ------------------------------------------------------ 9-way scatter-add
// One wave per point; lane covers 4 channels (b128 source load).
__global__ void scatter_kernel(const float4* __restrict__ f4, const int* __restrict__ seg,
                               float* __restrict__ out){
  int lane = threadIdx.x & 31, wave = threadIdx.x >> 5;
  int bn = blockIdx.x * 4 + wave;              // BB*NN/4 blocks of 128 threads
  int b = bn / NN;
  float4 v = f4[(long)bn*32 + lane];
  #pragma unroll
  for (int j = 0; j < 9; ++j){
    int s = seg[bn*9 + j];
    if (s >= 0 && s < NSEG){
      float* dst = &out[(((long)b*NSEG + s)*GD) + lane*4];
      atomAddF(dst + 0, v.x);
      atomAddF(dst + 1, v.y);
      atomAddF(dst + 2, v.z);
      atomAddF(dst + 3, v.w);
    }
  }
}

extern "C" void kernel_launch(void* const* d_in, const int* in_sizes, int n_in,
                              void* d_out, int out_size, void* d_ws, size_t ws_size,
                              hipStream_t stream){
  const float* opc   = (const float*)d_in[0];
  const float* pc    = (const float*)d_in[1];
  const float* w_in  = (const float*)d_in[2];
  const float* b_in  = (const float*)d_in[3];
  const float* w_g   = (const float*)d_in[4];
  const float* gamma = (const float*)d_in[5];
  const float* beta  = (const float*)d_in[6];
  const float* w_p   = (const float*)d_in[7];
  const float* b_p   = (const float*)d_in[8];
  float* out = (float*)d_out;
  (void)in_sizes; (void)n_in; (void)out_size; (void)ws_size;

  char* ws = (char*)d_ws;
  size_t off = 0;
  auto alloc = [&](size_t bytes) -> char* {
    char* p = ws + off;
    off = (off + bytes + 255) & ~(size_t)255;
    return p;
  };
  int*   seg  = (int*)  alloc((size_t)BB*NN*9*sizeof(int));
  float* x    = (float*)alloc((size_t)BB*NN*TD*sizeof(float));
  int*   idx  = (int*)  alloc((size_t)BB*NN*KK*sizeof(int));
  float* y    = (float*)alloc((size_t)BB*NN*GD*sizeof(float));
  float* base = (float*)alloc((size_t)BB*NN*GD*sizeof(float));
  float* mbuf = (float*)alloc((size_t)BB*NN*GD*sizeof(float));
  float* fbuf = (float*)alloc((size_t)BB*NN*GD*sizeof(float));
  float* stats= (float*)alloc((size_t)BB*8*sizeof(float));
  float* scale= (float*)alloc((size_t)BB*GD*sizeof(float));
  float* shift= (float*)alloc((size_t)BB*GD*sizeof(float));
  unsigned short* wfU = (unsigned short*)alloc((size_t)2*16*32*16*sizeof(unsigned short));
  unsigned short* wfP = (unsigned short*)alloc((size_t)4*8*32*16*sizeof(unsigned short));

  long n4 = (long)BB * NSEG * GD / 4;
  fill_zero_f4<<<4096, 256, 0, stream>>>((float4*)out, n4);
  prep_kernel<<<1, 256, 0, stream>>>(w_g, w_p, wfU, wfP, stats);
  seg_kernel<<<BB, 256, 0, stream>>>(pc, seg);
  intrans_kernel<<<BB*NN, TD, 0, stream>>>(opc, w_in, b_in, x);
  knn_kernel<<<BB*NN/8, 256, 0, stream>>>(opc, idx);
  gemm_yb_kernel<<<BB*NN/16, 512, 0, stream>>>(x, wfU, y, base);
  gather_max_kernel<<<BB*NN/4, 128, 0, stream>>>((const float4*)y, (const float4*)base,
                                                 idx, (float4*)mbuf, stats);
  finalize_kernel<<<2, 256, 0, stream>>>(stats, gamma, beta, scale, shift);
  norm_proj_kernel<<<BB*NN/16, 256, 0, stream>>>((const float4*)mbuf, (const float4*)scale,
                                                 (const float4*)shift, wfP, b_p, fbuf);
  scatter_kernel<<<BB*NN/4, 128, 0, stream>>>((const float4*)fbuf, seg, out);
}
